// GATNet_59811714564606
// MI455X (gfx1250) — matrix operand order repaired and checked by reading, not compiled, not run on hardware
//
#include <hip/hip_runtime.h>
#include <math.h>

typedef __attribute__((ext_vector_type(16))) __bf16       v16bf;
typedef __attribute__((ext_vector_type(8)))  float        v8f;
typedef __attribute__((ext_vector_type(8)))  unsigned int v8u;

#define NN 50000
#define MPAD 50048            // 391 * 128
#define NE 200000
#define ET (NE + NN)          // edges + self loops
#define NG 2048
#define NEG_SLOPE 0.2f
#define BNEPS 1e-5f

static inline int cdiv(long a, long b) { return (int)((a + b - 1) / b); }
static inline int rup(int a, int b) { return ((a + b - 1) / b) * b; }

// ---------------------------------------------------------------- utilities
__global__ void fill_f32(float* __restrict__ p, float v, long n) {
  long i = (long)blockIdx.x * blockDim.x + threadIdx.x;
  if (i < n) p[i] = v;
}

// activations: src[rows x K] (rows=NN valid) -> dst[MPAD x Kp], zero padded
__global__ void cvt_pad_act(const float* __restrict__ src, __bf16* __restrict__ dst,
                            int K, int Kp, long total) {
  long i = (long)blockIdx.x * blockDim.x + threadIdx.x;
  if (i >= total) return;
  int r = (int)(i / Kp), c = (int)(i % Kp);
  float v = (r < NN && c < K) ? src[(size_t)r * K + c] : 0.f;
  dst[i] = (__bf16)v;
}

// weights: W[K x N] -> dst transposed [1024 x Kp]: dst[n*Kp+k] = W[k*N+n], zero padded
__global__ void cvt_pad_wT(const float* __restrict__ W, __bf16* __restrict__ dst,
                           int K, int N, int Kp, long total) {
  long i = (long)blockIdx.x * blockDim.x + threadIdx.x;
  if (i >= total) return;
  int n = (int)(i / Kp), k = (int)(i % Kp);
  float v = (k < K && n < N) ? W[(size_t)k * N + n] : 0.f;
  dst[i] = (__bf16)v;
}

// ---------------------------------------------------------------- WMMA GEMM
// C[MPAD x Nld] = A[MPAD x Kp] * Bt[n][k]   (A row-major, B pre-transposed)
#define BM 128
#define BN 128
#define BK 32
#define KPAD 40               // 80-byte rows: 16B-aligned, bank-staggered

__global__ __launch_bounds__(256) void gemm_bf16_wmma(
    const __bf16* __restrict__ A, const __bf16* __restrict__ Bt,
    float* __restrict__ C, int Kp, int Nld)
{
  __shared__ __bf16 sA[BM][KPAD];
  __shared__ __bf16 sB[BN][KPAD];

  const int tid  = threadIdx.x;
  const int wave = tid >> 5;        // 0..7 (wave32)
  const int lane = tid & 31;
  const int wm   = wave & 1;        // 2 slabs of 64 rows
  const int wn   = wave >> 1;       // 4 slabs of 32 cols
  const int half = lane >> 4;
  const int l16  = lane & 15;

  const int bm = blockIdx.y * BM;
  const int bn = blockIdx.x * BN;

  v8f acc[4][2] = {};

  for (int k0 = 0; k0 < Kp; k0 += BK) {
    // ---- branch-free b128 staging: 512 uint4 per tile, 2 per thread
#pragma unroll
    for (int j = 0; j < 2; ++j) {
      int i = tid + j * 256;
      int r = i >> 2, m = (i & 3) * 8;
      *reinterpret_cast<uint4*>(&sA[r][m]) =
          *reinterpret_cast<const uint4*>(&A[(size_t)(bm + r) * Kp + k0 + m]);
      *reinterpret_cast<uint4*>(&sB[r][m]) =
          *reinterpret_cast<const uint4*>(&Bt[(size_t)(bn + r) * Kp + k0 + m]);
    }
    __syncthreads();

    // A 16x32 bf16 layout: lanes 0-15: K={0..7,16..23}; lanes 16-31: K={8..15,24..31}
    v16bf afrag[4], bfrag[2];
#pragma unroll
    for (int mt = 0; mt < 4; ++mt) {
      int row = wm * 64 + mt * 16 + l16;
      v8u raw;
#pragma unroll
      for (int d = 0; d < 8; ++d) {
        int kk = ((d < 4) ? 2 * d : 8 + 2 * d) + 8 * half;
        raw[d] = *reinterpret_cast<const unsigned int*>(&sA[row][kk]);
      }
      afrag[mt] = __builtin_bit_cast(v16bf, raw);
    }
    // B 32x16 layout: lanes 0-15: K={0..7,16..23}; lanes 16-31: K={8..15,24..31}, N=lane%16
#pragma unroll
    for (int nt = 0; nt < 2; ++nt) {
      int col = wn * 32 + nt * 16 + l16;
      v8u raw;
#pragma unroll
      for (int d = 0; d < 8; ++d) {
        int kk = ((d < 4) ? 2 * d : 8 + 2 * d) + 8 * half;
        raw[d] = *reinterpret_cast<const unsigned int*>(&sB[col][kk]);
      }
      bfrag[nt] = __builtin_bit_cast(v16bf, raw);
    }

#pragma unroll
    for (int mt = 0; mt < 4; ++mt)
#pragma unroll
      for (int nt = 0; nt < 2; ++nt)
        acc[mt][nt] = __builtin_amdgcn_wmma_f32_16x16x32_bf16(
            false, afrag[mt], false, bfrag[nt], (short)0, acc[mt][nt], false, false);
    __syncthreads();
  }

  // C/D 16x16 f32 layout: lane -> N, vgpr v -> M = v + 8*half
#pragma unroll
  for (int mt = 0; mt < 4; ++mt)
#pragma unroll
    for (int nt = 0; nt < 2; ++nt) {
      int c = bn + wn * 32 + nt * 16 + l16;
      if (c < Nld) {
#pragma unroll
        for (int v = 0; v < 8; ++v) {
          int r = bm + wm * 64 + mt * 16 + v + 8 * half;
          C[(size_t)r * Nld + c] = acc[mt][nt][v];
        }
      }
    }
}

// ---------------------------------------------------------------- GAT pieces
__global__ void node_scalars(const float* __restrict__ H,
                             const float* __restrict__ as, const float* __restrict__ ad,
                             float* __restrict__ hs, float* __restrict__ hd,
                             int D, int Hld)
{
  int node = blockIdx.x * (blockDim.x >> 5) + (threadIdx.x >> 5);
  if (node >= NN) return;
  int lane = threadIdx.x & 31;
  float s = 0.f, d = 0.f;
  const float* row = H + (size_t)node * Hld;
  for (int c = lane; c < D; c += 32) {
    float h = row[c];
    s += h * as[c];
    d += h * ad[c];
  }
  for (int o = 16; o > 0; o >>= 1) {
    s += __shfl_xor(s, o, 32);
    d += __shfl_xor(d, o, 32);
  }
  if (lane == 0) { hs[node] = s; hd[node] = d; }
}

__device__ __forceinline__ void edge_sd(const int* ei, int i, int& s, int& d) {
  if (i < NE) { s = ei[i]; d = ei[NE + i]; }
  else        { s = d = i - NE; }            // self loops
}

__global__ void edge_max(const int* __restrict__ ei,
                         const float* __restrict__ hs, const float* __restrict__ hd,
                         float* __restrict__ ebuf, float* __restrict__ emax)
{
  int i = blockIdx.x * blockDim.x + threadIdx.x;
  if (i >= ET) return;
  int s, d; edge_sd(ei, i, s, d);
  float e = hs[s] + hd[d];
  e = (e > 0.f) ? e : NEG_SLOPE * e;         // leaky_relu
  ebuf[i] = e;
  __hip_atomic_fetch_max(&emax[d], e, __ATOMIC_RELAXED, __HIP_MEMORY_SCOPE_AGENT);
}

__global__ void edge_expsum(const int* __restrict__ ei,
                            const float* __restrict__ emax,
                            float* __restrict__ ebuf, float* __restrict__ denom)
{
  int i = blockIdx.x * blockDim.x + threadIdx.x;
  if (i >= ET) return;
  int s, d; edge_sd(ei, i, s, d);
  float ex = expf(ebuf[i] - emax[d]);
  ebuf[i] = ex;                               // reuse buffer for ex
  __hip_atomic_fetch_add(&denom[d], ex, __ATOMIC_RELAXED, __HIP_MEMORY_SCOPE_AGENT);
}

// one block per edge; float4 gathers; agg[dst] += h[src]*alpha
__global__ void edge_scatter(const int* __restrict__ ei,
                             const float* __restrict__ ebuf, const float* __restrict__ denom,
                             const float* __restrict__ H, float* __restrict__ AGG,
                             int D, int Hld)
{
  int i = blockIdx.x;
  if (i >= ET) return;
  int s, d; edge_sd(ei, i, s, d);
  float alpha = ebuf[i] / denom[d];
  const float4* hr4 = reinterpret_cast<const float4*>(H + (size_t)s * Hld);
  float* arow = AGG + (size_t)d * D;
  int D4 = D >> 2;
  for (int c = threadIdx.x; c < D4; c += blockDim.x) {
    float4 h4 = hr4[c];
    int b = c * 4;
    __hip_atomic_fetch_add(&arow[b + 0], h4.x * alpha, __ATOMIC_RELAXED, __HIP_MEMORY_SCOPE_AGENT);
    __hip_atomic_fetch_add(&arow[b + 1], h4.y * alpha, __ATOMIC_RELAXED, __HIP_MEMORY_SCOPE_AGENT);
    __hip_atomic_fetch_add(&arow[b + 2], h4.z * alpha, __ATOMIC_RELAXED, __HIP_MEMORY_SCOPE_AGENT);
    __hip_atomic_fetch_add(&arow[b + 3], h4.w * alpha, __ATOMIC_RELAXED, __HIP_MEMORY_SCOPE_AGENT);
  }
}

// x = bn(relu(agg + b)) in place
__global__ void bias_relu_bn(float* __restrict__ X, const float* __restrict__ b,
                             const float* __restrict__ g, const float* __restrict__ be,
                             const float* __restrict__ rm, const float* __restrict__ rv,
                             int D, long n)
{
  long i = (long)blockIdx.x * blockDim.x + threadIdx.x;
  if (i >= n) return;
  int c = (int)(i % D);
  float v = X[i] + b[c];
  v = fmaxf(v, 0.f);
  v = (v - rm[c]) * rsqrtf(rv[c] + BNEPS) * g[c] + be[c];
  X[i] = v;
}

// global add pool: one block per node, float4 reads
__global__ void pool_add(const float* __restrict__ X, const int* __restrict__ batch,
                         float* __restrict__ G, int D)
{
  int n = blockIdx.x;
  if (n >= NN) return;
  int gidx = batch[n];
  const float4* r4 = reinterpret_cast<const float4*>(X + (size_t)n * D);
  float* grow = G + (size_t)gidx * D;
  int D4 = D >> 2;
  for (int c = threadIdx.x; c < D4; c += blockDim.x) {
    float4 v = r4[c];
    int b = c * 4;
    __hip_atomic_fetch_add(&grow[b + 0], v.x, __ATOMIC_RELAXED, __HIP_MEMORY_SCOPE_AGENT);
    __hip_atomic_fetch_add(&grow[b + 1], v.y, __ATOMIC_RELAXED, __HIP_MEMORY_SCOPE_AGENT);
    __hip_atomic_fetch_add(&grow[b + 2], v.z, __ATOMIC_RELAXED, __HIP_MEMORY_SCOPE_AGENT);
    __hip_atomic_fetch_add(&grow[b + 3], v.w, __ATOMIC_RELAXED, __HIP_MEMORY_SCOPE_AGENT);
  }
}

// small dense fc: Y[M,N] = act(X[M,K] @ W[K,N] + b)
__global__ void fc_naive(const float* __restrict__ X, const float* __restrict__ W,
                         const float* __restrict__ b, float* __restrict__ Y,
                         int M, int K, int N, int relu)
{
  int i = blockIdx.x * blockDim.x + threadIdx.x;
  if (i >= M * N) return;
  int m = i / N, n = i % N;
  const float* xr = X + (size_t)m * K;
  float acc = b[n];
  for (int k = 0; k < K; ++k) acc += xr[k] * W[(size_t)k * N + n];
  if (relu) acc = fmaxf(acc, 0.f);
  Y[i] = acc;
}

// ---------------------------------------------------------------- driver
extern "C" void kernel_launch(void* const* d_in, const int* in_sizes, int n_in,
                              void* d_out, int out_size, void* d_ws, size_t ws_size,
                              hipStream_t stream)
{
  (void)in_sizes; (void)n_in; (void)out_size; (void)ws_size;

  const float* x0    = (const float*)d_in[0];
  const int*   ei    = (const int*)d_in[1];
  const int*   batch = (const int*)d_in[2];

  const int DI[5] = {29, 1024, 512, 256, 516};
  const int DO[5] = {1024, 512, 256, 516, 1024};

  // --- workspace layout (bump allocator)
  char* ws = (char*)d_ws;
  size_t off = 0;
  auto alloc = [&](size_t bytes) {
    off = (off + 255) & ~(size_t)255;
    void* p = ws + off;
    off += bytes;
    return p;
  };
  __bf16* Abf   = (__bf16*)alloc((size_t)MPAD * 1024 * 2);
  __bf16* WbfT  = (__bf16*)alloc((size_t)1024 * 1024 * 2);
  float*  H     = (float*) alloc((size_t)MPAD * 1024 * 4);
  float*  X     = (float*) alloc((size_t)NN * 1024 * 4);  // prev act / current agg
  float*  hs    = (float*) alloc((size_t)NN * 4);
  float*  hd    = (float*) alloc((size_t)NN * 4);
  float*  emax  = (float*) alloc((size_t)NN * 4);
  float*  denom = (float*) alloc((size_t)NN * 4);
  float*  ebuf  = (float*) alloc((size_t)ET * 4);
  float*  G     = (float*) alloc((size_t)NG * 1024 * 4);
  float*  F1    = (float*) alloc((size_t)NG * 128 * 4);
  float*  F2    = (float*) alloc((size_t)NG * 16 * 4);

  for (int l = 0; l < 5; ++l) {
    const int di = DI[l], dn = DO[l];
    const int Kp  = rup(di, BK);     // 32, 1024, 512, 256, 544
    const int Nld = rup(dn, 64);     // 1024, 512, 256, 576, 1024
    const float* Wl  = (const float*)d_in[3 + 4 * l + 0];
    const float* asv = (const float*)d_in[3 + 4 * l + 1];
    const float* adv = (const float*)d_in[3 + 4 * l + 2];
    const float* bl  = (const float*)d_in[3 + 4 * l + 3];
    const float* bg  = (const float*)d_in[23 + 4 * l + 0];
    const float* bbe = (const float*)d_in[23 + 4 * l + 1];
    const float* brm = (const float*)d_in[23 + 4 * l + 2];
    const float* brv = (const float*)d_in[23 + 4 * l + 3];

    const float* xin = (l == 0) ? x0 : X;

    // 1) padded bf16 copies (A row-major, W transposed)
    long na = (long)MPAD * Kp;
    cvt_pad_act<<<cdiv(na, 256), 256, 0, stream>>>(xin, Abf, di, Kp, na);
    long nw = (long)1024 * Kp;
    cvt_pad_wT<<<cdiv(nw, 256), 256, 0, stream>>>(Wl, WbfT, di, dn, Kp, nw);

    // 2) H = A @ W  (8x v_wmma per K-step, branch-free b128 staging)
    dim3 ggrid(cdiv(Nld, BN), MPAD / BM);
    gemm_bf16_wmma<<<ggrid, 256, 0, stream>>>(Abf, WbfT, H, Kp, Nld);

    // 3) attention scalars
    node_scalars<<<cdiv(NN, 8), 256, 0, stream>>>(H, asv, adv, hs, hd, dn, Nld);

    // 4) init reductions (X becomes the aggregation target now)
    fill_f32<<<cdiv(NN, 256), 256, 0, stream>>>(emax, -3.4028235e38f, NN);
    fill_f32<<<cdiv(NN, 256), 256, 0, stream>>>(denom, 0.f, NN);
    long nagg = (long)NN * dn;
    fill_f32<<<cdiv(nagg, 256), 256, 0, stream>>>(X, 0.f, nagg);

    // 5) segment softmax over incoming edges + scatter
    edge_max   <<<cdiv(ET, 256), 256, 0, stream>>>(ei, hs, hd, ebuf, emax);
    edge_expsum<<<cdiv(ET, 256), 256, 0, stream>>>(ei, emax, ebuf, denom);
    edge_scatter<<<ET, 256, 0, stream>>>(ei, ebuf, denom, H, X, dn, Nld);

    // 6) bias + relu + inference BN (in place)
    bias_relu_bn<<<cdiv(nagg, 256), 256, 0, stream>>>(X, bl, bg, bbe, brm, brv, dn, nagg);
  }

  // ---- global_add_pool
  long ng = (long)NG * 1024;
  fill_f32<<<cdiv(ng, 256), 256, 0, stream>>>(G, 0.f, ng);
  pool_add<<<NN, 256, 0, stream>>>(X, batch, G, 1024);

  // ---- FC head
  const float* w2 = (const float*)d_in[43];
  const float* b2 = (const float*)d_in[44];
  const float* w3 = (const float*)d_in[45];
  const float* b3 = (const float*)d_in[46];
  const float* w4 = (const float*)d_in[47];
  const float* b4 = (const float*)d_in[48];

  fc_naive<<<cdiv((long)NG * 128, 256), 256, 0, stream>>>(G,  w2, b2, F1, NG, 1024, 128, 1);
  fc_naive<<<cdiv((long)NG * 16, 256), 256, 0, stream>>>(F1, w3, b3, F2, NG, 128, 16, 1);
  fc_naive<<<cdiv((long)NG * 1, 256), 256, 0, stream>>>(F2, w4, b4, (float*)d_out, NG, 16, 1, 0);
}
